// ModulatedDeformConv2d_44418551775255
// MI455X (gfx1250) — matrix operand order repaired and checked by reading, not compiled, not run on hardware
//
#include <hip/hip_runtime.h>
#include <hip/hip_bf16.h>
#include <math.h>

typedef __attribute__((ext_vector_type(16))) _Float16 v16h;
typedef __attribute__((ext_vector_type(8)))  _Float16 v8h;
typedef __attribute__((ext_vector_type(8)))  float    v8f;

#define NB 4
#define CIN 64
#define COUT 64
#define HH 128
#define WW 128
#define K2 9
#define VSTRIDE 72   // halfs per pixel row in LDS (144B, 16B aligned)

// ---------------------------------------------------------------------------
// Prepack main-GEMM weights (Cout=64, Cin=64, 9) into WMMA A-fragment order.
// Layout: [tap 9][stripe 4][chunk 2][lane 32][elem 16] f16, each lane's 16
// halfs contiguous (one 32B load per lane).
// A 16-bit 16x32 fragment: lane<16 -> K = e (e<8) else e+8 ; lane>=16 -> +8/+16.
// ---------------------------------------------------------------------------
__global__ void prepackA3(const float* __restrict__ w, _Float16* __restrict__ Apack) {
    int idx = blockIdx.x * 256 + threadIdx.x;
    if (idx >= 9 * 4 * 2 * 32 * 16) return;
    int e      = idx & 15;
    int lane   = (idx >> 4) & 31;
    int chunk  = (idx >> 9) & 1;
    int stripe = (idx >> 10) & 3;
    int k      = idx >> 12;
    int M = stripe * 16 + (lane & 15);
    int klo = (lane < 16) ? ((e < 8) ? e : e + 8)
                          : ((e < 8) ? e + 8 : e + 16);
    int Kd = chunk * 32 + klo;                 // tap-local input channel
    Apack[idx] = (_Float16)w[(M * CIN + Kd) * K2 + k];
}

// Prepack offset+mask conv weights: rows 0..17 = w_off, 18..26 = w_mask, 27..31 = 0.
// Layout: [tap 9][stripe 2][chunk 2][lane 32][elem 16] f16.
__global__ void prepackA1(const float* __restrict__ w_off, const float* __restrict__ w_mask,
                          _Float16* __restrict__ Apack) {
    int idx = blockIdx.x * 256 + threadIdx.x;
    if (idx >= 9 * 2 * 2 * 32 * 16) return;
    int e      = idx & 15;
    int lane   = (idx >> 4) & 31;
    int chunk  = (idx >> 9) & 1;
    int stripe = (idx >> 10) & 1;
    int k      = idx >> 11;
    int M = stripe * 16 + (lane & 15);
    int klo = (lane < 16) ? ((e < 8) ? e : e + 8)
                          : ((e < 8) ? e + 8 : e + 16);
    int Kd = chunk * 32 + klo;
    float v = 0.f;
    if (M < 18)       v = w_off[(M * CIN + Kd) * K2 + k];
    else if (M < 27)  v = w_mask[((M - 18) * CIN + Kd) * K2 + k];
    Apack[idx] = (_Float16)v;
}

// ---------------------------------------------------------------------------
// Stage 1: offset(18ch)+mask(9ch) 3x3 conv via WMMA implicit GEMM, then
// offmod[ch] = (conv_off[ch]+b_off) * sigmoid(conv_mask[ch%9]+b_mask).
// One block per (n, ho): 256 threads = 8 waves. Output 32 rows x 128 px.
// ---------------------------------------------------------------------------
__global__ void __launch_bounds__(256)
stage1_offset(const float* __restrict__ x, const float* __restrict__ b_off,
              const float* __restrict__ b_mask, const _Float16* __restrict__ Apack1,
              float* __restrict__ offmod) {
    __shared__ _Float16 val[WW * VSTRIDE];     // B tile [pixel][chan]
    __shared__ float    accf[32 * 132];        // conv results for epilogue

    const int n  = blockIdx.x >> 7;
    const int ho = blockIdx.x & 127;
    const int tid  = threadIdx.x;
    const int lane = tid & 31;
    const int wv   = tid >> 5;
    const int stripe = wv & 1;                 // output-row stripe (0..1)
    const int pt0    = wv >> 1;                // pixel tiles pt0 and pt0+4

    v8f acc0, acc1;
    #pragma unroll
    for (int r = 0; r < 8; ++r) { acc0[r] = 0.f; acc1[r] = 0.f; }

    const float* xn = x + (size_t)n * CIN * HH * WW;

    for (int k = 0; k < K2; ++k) {
        const int ky = k / 3, kx = k % 3;
        const int y = ho + ky - 1;
        // Fill B tile: val[p][c] = x[n][c][y][p+kx-1] (zero padded), f16.
        for (int idx = tid; idx < WW * CIN; idx += 256) {
            int p = idx & 127;
            int c = idx >> 7;
            int xr = p + kx - 1;
            float v = 0.f;
            if ((unsigned)y < (unsigned)HH && (unsigned)xr < (unsigned)WW)
                v = xn[(c * HH + y) * WW + xr];
            val[p * VSTRIDE + c] = (_Float16)v;
        }
        __syncthreads();

        const v16h a0 = *(const v16h*)(Apack1 + (((k * 2 + stripe) * 2 + 0) * 32 + lane) * 16);
        const v16h a1 = *(const v16h*)(Apack1 + (((k * 2 + stripe) * 2 + 1) * 32 + lane) * 16);

        const int cb = (lane >> 4) << 3;       // lane<16 -> K 0..7,16..23, lane>=16 -> 8..15,24..31
        #pragma unroll
        for (int t = 0; t < 2; ++t) {
            int ptile = pt0 + t * 4;
            int pl = ptile * 16 + (lane & 15);
            union { v16h v; v8h h[2]; } b0, b1;
            b0.h[0] = *(const v8h*)(&val[pl * VSTRIDE + cb]);
            b0.h[1] = *(const v8h*)(&val[pl * VSTRIDE + cb + 16]);
            b1.h[0] = *(const v8h*)(&val[pl * VSTRIDE + 32 + cb]);
            b1.h[1] = *(const v8h*)(&val[pl * VSTRIDE + 32 + cb + 16]);
            if (t == 0) {
                acc0 = __builtin_amdgcn_wmma_f32_16x16x32_f16(false, a0, false, b0.v, (short)0, acc0, false, false);
                acc0 = __builtin_amdgcn_wmma_f32_16x16x32_f16(false, a1, false, b1.v, (short)0, acc0, false, false);
            } else {
                acc1 = __builtin_amdgcn_wmma_f32_16x16x32_f16(false, a0, false, b0.v, (short)0, acc1, false, false);
                acc1 = __builtin_amdgcn_wmma_f32_16x16x32_f16(false, a1, false, b1.v, (short)0, acc1, false, false);
            }
        }
        __syncthreads();
    }

    // Spill accumulators (C/D layout: elem r -> row r + 8*(lane>=16), col lane&15).
    {
        int col = pt0 * 16 + (lane & 15);
        int rb  = stripe * 16 + ((lane >> 4) << 3);
        #pragma unroll
        for (int r = 0; r < 8; ++r) {
            accf[(rb + r) * 132 + col]      = acc0[r];
            accf[(rb + r) * 132 + col + 64] = acc1[r];
        }
    }
    __syncthreads();

    for (int idx = tid; idx < 18 * WW; idx += 256) {
        int j = idx >> 7;
        int p = idx & 127;
        int jm = j % 9;
        float off = accf[j * 132 + p] + b_off[j];
        float m   = accf[(18 + jm) * 132 + p] + b_mask[jm];
        float sig = 1.f / (1.f + __expf(-m));
        offmod[(((size_t)n * 18 + j) * HH + ho) * WW + p] = off * sig;
    }
}

// ---------------------------------------------------------------------------
// Stage 2: fused bilinear gather + main GEMM (Cout=64 x K=576 x 128 px/row).
// One block per (n, ho). Per tap: build val[p][c] f16 in LDS, then 8 waves do
// (stripe = wv&3, pixel half = wv>>2) x 4 tiles x 2 K-chunks of WMMA.
// ---------------------------------------------------------------------------
__global__ void __launch_bounds__(256)
stage2_deform(const float* __restrict__ x, const float* __restrict__ offmod,
              const _Float16* __restrict__ Apack3, const float* __restrict__ bias,
              float* __restrict__ out) {
    __shared__ float    offb[18 * WW];
    __shared__ _Float16 val[WW * VSTRIDE];

    const int n  = blockIdx.x >> 7;
    const int ho = blockIdx.x & 127;
    const int tid  = threadIdx.x;
    const int lane = tid & 31;
    const int wv   = tid >> 5;
    const int stripe = wv & 3;                 // Cout rows [16*stripe, +16)
    const int half   = wv >> 2;                // pixels [64*half, +64)

    for (int idx = tid; idx < 18 * WW; idx += 256)
        offb[idx] = offmod[(((size_t)n * 18 + (idx >> 7)) * HH + ho) * WW + (idx & 127)];
    __syncthreads();

    v8f acc[4];
    #pragma unroll
    for (int t = 0; t < 4; ++t)
        #pragma unroll
        for (int r = 0; r < 8; ++r) acc[t][r] = 0.f;

    const float* xn = x + (size_t)n * CIN * HH * WW;
    const int p2    = tid & 127;               // sampling pixel for this thread
    const int chalf = (tid >> 7) & 1;          // channels [32*chalf, +32)

    for (int k = 0; k < K2; ++k) {
        const int ki = k / 3, kj = k % 3;
        // --- bilinear sample 32 channels at this thread's pixel/tap ---
        float offy = offb[(2 * k) * WW + p2];
        float offx = offb[(2 * k + 1) * WW + p2];
        float py = (float)(ho - 1 + ki) + offy;
        float px = (float)(p2 - 1 + kj) + offx;
        float y0f = floorf(py), x0f = floorf(px);
        float wy = py - y0f, wx = px - x0f;
        int y0 = (int)y0f, x0 = (int)x0f;
        int y1 = y0 + 1, x1 = x0 + 1;
        float w00 = (1.f - wy) * (1.f - wx);
        float w01 = (1.f - wy) * wx;
        float w10 = wy * (1.f - wx);
        float w11 = wy * wx;
        bool vy0 = (unsigned)y0 < (unsigned)HH, vy1 = (unsigned)y1 < (unsigned)HH;
        bool vx0 = (unsigned)x0 < (unsigned)WW, vx1 = (unsigned)x1 < (unsigned)WW;
        if (!(vy0 && vx0)) w00 = 0.f;
        if (!(vy0 && vx1)) w01 = 0.f;
        if (!(vy1 && vx0)) w10 = 0.f;
        if (!(vy1 && vx1)) w11 = 0.f;
        int yc0 = min(max(y0, 0), HH - 1), yc1 = min(max(y1, 0), HH - 1);
        int xc0 = min(max(x0, 0), WW - 1), xc1 = min(max(x1, 0), WW - 1);
        int i00 = yc0 * WW + xc0, i01 = yc0 * WW + xc1;
        int i10 = yc1 * WW + xc0, i11 = yc1 * WW + xc1;
        const float* xc = xn + (size_t)chalf * 32 * HH * WW;
        for (int c = 0; c < 32; ++c) {
            const float* b = xc + (size_t)c * HH * WW;
            float v = w00 * b[i00] + w01 * b[i01] + w10 * b[i10] + w11 * b[i11];
            val[p2 * VSTRIDE + chalf * 32 + c] = (_Float16)v;
        }
        __syncthreads();

        // --- WMMA accumulate this tap's K=64 (two 16x16x32 chunks) ---
        const v16h a0 = *(const v16h*)(Apack3 + (((k * 4 + stripe) * 2 + 0) * 32 + lane) * 16);
        const v16h a1 = *(const v16h*)(Apack3 + (((k * 4 + stripe) * 2 + 1) * 32 + lane) * 16);
        const int cb = (lane >> 4) << 3;
        #pragma unroll
        for (int t = 0; t < 4; ++t) {
            int pl = (half * 4 + t) * 16 + (lane & 15);
            union { v16h v; v8h h[2]; } b0, b1;
            b0.h[0] = *(const v8h*)(&val[pl * VSTRIDE + cb]);
            b0.h[1] = *(const v8h*)(&val[pl * VSTRIDE + cb + 16]);
            b1.h[0] = *(const v8h*)(&val[pl * VSTRIDE + 32 + cb]);
            b1.h[1] = *(const v8h*)(&val[pl * VSTRIDE + 32 + cb + 16]);
            acc[t] = __builtin_amdgcn_wmma_f32_16x16x32_f16(false, a0, false, b0.v, (short)0, acc[t], false, false);
            acc[t] = __builtin_amdgcn_wmma_f32_16x16x32_f16(false, a1, false, b1.v, (short)0, acc[t], false, false);
        }
        __syncthreads();
    }

    // Epilogue: out[n][o][ho][p] = acc + bias[o]
    const int colb = half * 64 + (lane & 15);
    const int rb   = stripe * 16 + ((lane >> 4) << 3);
    #pragma unroll
    for (int t = 0; t < 4; ++t) {
        int p = colb + t * 16;
        #pragma unroll
        for (int r = 0; r < 8; ++r) {
            int o = rb + r;
            out[(((size_t)n * COUT + o) * HH + ho) * WW + p] = acc[t][r] + bias[o];
        }
    }
}

// ---------------------------------------------------------------------------
extern "C" void kernel_launch(void* const* d_in, const int* in_sizes, int n_in,
                              void* d_out, int out_size, void* d_ws, size_t ws_size,
                              hipStream_t stream) {
    const float* x      = (const float*)d_in[0];
    const float* w_off  = (const float*)d_in[1];
    const float* b_off  = (const float*)d_in[2];
    const float* w_mask = (const float*)d_in[3];
    const float* b_mask = (const float*)d_in[4];
    const float* weight = (const float*)d_in[5];
    const float* bias   = (const float*)d_in[6];
    float* out = (float*)d_out;

    // Workspace partition (256B-aligned offsets)
    float*    offmod = (float*)d_ws;                                   // 4,718,592 B
    _Float16* Apack3 = (_Float16*)((char*)d_ws + 4718592);             //    73,728 B
    _Float16* Apack1 = (_Float16*)((char*)d_ws + 4718592 + 73728);     //    36,864 B

    prepackA3<<<(9 * 4 * 2 * 32 * 16 + 255) / 256, 256, 0, stream>>>(weight, Apack3);
    prepackA1<<<(9 * 2 * 2 * 32 * 16 + 255) / 256, 256, 0, stream>>>(w_off, w_mask, Apack1);

    dim3 grid(NB * HH);   // one block per (n, ho)
    stage1_offset<<<grid, 256, 0, stream>>>(x, b_off, b_mask, Apack1, offmod);
    stage2_deform<<<grid, 256, 0, stream>>>(x, offmod, Apack3, bias, out);
}
